// _3d_deform_1228360646682
// MI455X (gfx1250) — matrix operand order repaired and checked by reading, not compiled, not run on hardware
//
#include <hip/hip_runtime.h>

// ---------------------------------------------------------------------------
// Problem constants (from reference): B=2, C=64, D=16, H=32, W=32, K=27 taps
// ---------------------------------------------------------------------------
#define CB   64
#define DD   16
#define HH   32
#define WW   32
#define VOX  16384          // D*H*W
#define NB   2
#define KTAP 27
#define KOFF 81             // 3*K offset channels
#define KOFFP 96            // padded to multiple of 16 for WMMA
#define NPAD 68             // f32 LDS row pitch (k_offconv raw tile)
#define BP   72             // ushort LDS row pitch (144B; 16B-aligned rows)

typedef __attribute__((ext_vector_type(16))) __bf16          v16bf;
typedef __attribute__((ext_vector_type(8)))  float           v8f;
typedef __attribute__((ext_vector_type(4)))  unsigned        u32x4;
typedef __attribute__((ext_vector_type(4)))  unsigned short  u16x4;
typedef __attribute__((ext_vector_type(8)))  unsigned short  u16x8;

union BF16x16 { v16bf v; unsigned short s[16]; };

__device__ __forceinline__ v8f wmma_bf16(v16bf a, v16bf b, v8f c) {
    return __builtin_amdgcn_wmma_f32_16x16x32_bf16(false, a, false, b, (short)0, c,
                                                   false, false);
}

// split fp32 -> bf16 hi + bf16 lo (round-to-nearest-even), a ~ hi + lo
__device__ __forceinline__ void bfsplit(float f, unsigned short& h, unsigned short& l) {
    unsigned u  = __float_as_uint(f);
    unsigned short hb = (unsigned short)((u + (0x7FFFu + ((u >> 16) & 1u))) >> 16);
    float hf = __uint_as_float((unsigned)hb << 16);
    float r  = f - hf;
    unsigned u2 = __float_as_uint(r);
    l = (unsigned short)((u2 + (0x7FFFu + ((u2 >> 16) & 1u))) >> 16);
    h = hb;
}

// A-fragment K map (contiguous, matching B): lanes 0-15 elems 0..15 -> K 0..15,
// lanes 16-31 elems 0..15 -> K 16..31.  Used only at weight-prep time to
// pre-permute weights into fragment element order.
__device__ __forceinline__ int kAmap(int j, int halfSel) {
    int k = j;
    return k + (halfSel ? 16 : 0);
}

// Load a 32B fragment (16 bf16 in element order) from LDS or global: two
// 16B vector loads, zero shuffle VALU.
__device__ __forceinline__ v16bf ldFrag16(const unsigned short* p) {
    union { v16bf v; u32x4 q[2]; } t;
    const u32x4* s = (const u32x4*)p;
    t.q[0] = s[0];
    t.q[1] = s[1];
    return t.v;
}

// ---------------------------------------------------------------------------
// K0: transpose x (B,C,D,H,W) -> voxel-major x_t (B,V,C)
// ---------------------------------------------------------------------------
__global__ __launch_bounds__(256) void k_transpose(const float* __restrict__ x,
                                                   float* __restrict__ xt) {
    int tid = blockIdx.x * 256 + threadIdx.x;
    int c = tid & 63;
    int lin = tid >> 6;
    int v = lin & (VOX - 1);
    int b = lin >> 14;
    xt[tid] = x[((size_t)(b * CB + c)) * VOX + v];
}

// ---------------------------------------------------------------------------
// K1: depthwise 5x5x5, pad 2 (voxel-major in/out)
// ---------------------------------------------------------------------------
__global__ __launch_bounds__(256) void k_dw5(const float* __restrict__ xt,
                                             const float* __restrict__ w0,
                                             const float* __restrict__ b0,
                                             float* __restrict__ t0) {
    int tid = blockIdx.x * 256 + threadIdx.x;
    int c = tid & 63;
    int lin = tid >> 6;
    int v = lin & (VOX - 1);
    int b = lin >> 14;
    int d = v >> 10, h = (v >> 5) & 31, w = v & 31;
    float s = b0[c];
    const float* wc = w0 + c * 125;
    for (int kd = 0; kd < 5; ++kd) {
        int dd = d + kd - 2;
        if ((unsigned)dd >= (unsigned)DD) continue;
        for (int kh = 0; kh < 5; ++kh) {
            int hh = h + kh - 2;
            if ((unsigned)hh >= (unsigned)HH) continue;
            for (int kw = 0; kw < 5; ++kw) {
                int ww = w + kw - 2;
                if ((unsigned)ww >= (unsigned)WW) continue;
                int nv = (dd << 10) + (hh << 5) + ww;
                s += xt[((size_t)(b * VOX + nv)) * CB + c] * wc[kd * 25 + kh * 5 + kw];
            }
        }
    }
    t0[tid] = s;
}

// ---------------------------------------------------------------------------
// K2: depthwise 7x7x7, dilation 3, pad 9 (voxel-major in/out)
// ---------------------------------------------------------------------------
__global__ __launch_bounds__(256) void k_dw7d3(const float* __restrict__ t0,
                                               const float* __restrict__ wsW,
                                               const float* __restrict__ bs,
                                               float* __restrict__ t1) {
    int tid = blockIdx.x * 256 + threadIdx.x;
    int c = tid & 63;
    int lin = tid >> 6;
    int v = lin & (VOX - 1);
    int b = lin >> 14;
    int d = v >> 10, h = (v >> 5) & 31, w = v & 31;
    float s = bs[c];
    const float* wc = wsW + c * 343;
    for (int kd = 0; kd < 7; ++kd) {
        int dd = d + kd * 3 - 9;
        if ((unsigned)dd >= (unsigned)DD) continue;
        for (int kh = 0; kh < 7; ++kh) {
            int hh = h + kh * 3 - 9;
            if ((unsigned)hh >= (unsigned)HH) continue;
            for (int kw = 0; kw < 7; ++kw) {
                int ww = w + kw * 3 - 9;
                if ((unsigned)ww >= (unsigned)WW) continue;
                int nv = (dd << 10) + (hh << 5) + ww;
                s += t0[((size_t)(b * VOX + nv)) * CB + c] * wc[kd * 49 + kh * 7 + kw];
            }
        }
    }
    t1[tid] = s;
}

// ---------------------------------------------------------------------------
// K3: split weights into hi/lo bf16, PRE-PERMUTED to A-fragment element order:
//   layout [tap][q][half][row][j], c = q*32 + kAmap(j, half)  (bijection on c)
//   w_off (81,64,27) -> rows = 96 (zero-padded)   (165888 elems)
//   w_def (64,64,27) -> rows = 64                 (110592 elems)
//   w1    (64,64)    -> rows = 64, no tap         (4096 elems)
// ---------------------------------------------------------------------------
#define WOFF_N (KTAP * 2 * 2 * KOFFP * 16)
#define WDEF_N (KTAP * 2 * 2 * CB * 16)
#define W1_N   (2 * 2 * CB * 16)
__global__ __launch_bounds__(256) void k_wprep(const float* __restrict__ w_off,
                                               const float* __restrict__ w_def,
                                               const float* __restrict__ w1,
                                               unsigned short* __restrict__ woAH,
                                               unsigned short* __restrict__ woAL,
                                               unsigned short* __restrict__ wdAH,
                                               unsigned short* __restrict__ wdAL,
                                               unsigned short* __restrict__ w1AH,
                                               unsigned short* __restrict__ w1AL) {
    int tid = blockIdx.x * 256 + threadIdx.x;
    if (tid < WOFF_N) {
        int j = tid & 15;
        int t2 = tid >> 4;
        int kk = t2 % KOFFP;
        int t3 = t2 / KOFFP;
        int half = t3 & 1, q = (t3 >> 1) & 1, tap = t3 >> 2;
        int c = q * 32 + kAmap(j, half);
        float f = 0.f;
        if (kk < KOFF) f = w_off[((size_t)(kk * CB + c)) * KTAP + tap];
        bfsplit(f, woAH[tid], woAL[tid]);
    } else if (tid < WOFF_N + WDEF_N) {
        int t = tid - WOFF_N;
        int j = t & 15;
        int t2 = t >> 4;
        int o = t2 & 63;
        int t3 = t2 >> 6;
        int half = t3 & 1, q = (t3 >> 1) & 1, tap = t3 >> 2;
        int c = q * 32 + kAmap(j, half);
        bfsplit(w_def[((size_t)(o * CB + c)) * KTAP + tap], wdAH[t], wdAL[t]);
    } else if (tid < WOFF_N + WDEF_N + W1_N) {
        int t = tid - WOFF_N - WDEF_N;
        int j = t & 15;
        int t2 = t >> 4;
        int o = t2 & 63;
        int t3 = t2 >> 6;
        int half = t3 & 1, q = t3 >> 1;
        int c = q * 32 + kAmap(j, half);
        bfsplit(w1[(size_t)o * CB + c], w1AH[t], w1AL[t]);
    }
}

// ---------------------------------------------------------------------------
// K4: offset conv = fused im2col + WMMA GEMM (M=96 padded, N=64/block).
//   im2col tile staged with GLOBAL_LOAD_ASYNC_TO_LDS_B128 (ASYNCcnt);
//   halo rows zero-filled with DS stores (DScnt); both fenced before WMMA.
//   A-fragments: direct 2x16B loads of pre-permuted bf16 weights.
// ---------------------------------------------------------------------------
__global__ __launch_bounds__(256) void k_offconv(const float* __restrict__ t1,
                                                 const unsigned short* __restrict__ wAH,
                                                 const unsigned short* __restrict__ wAL,
                                                 const float* __restrict__ boff,
                                                 float* __restrict__ offOut) {
    __shared__ float sBf[CB * NPAD];      // [n][c] raw f32 tile, pitch 68
    int tid = threadIdx.x;
    int lane = tid & 31, wv = tid >> 5;
    int halfSel = (lane >> 4) & 1;
    int lx = lane & 15;
    int b = blockIdx.x >> 8;
    int vbase = (blockIdx.x & 255) << 6;
    int ni = wv & 3;
    int nloc = ni * 16 + lx;

    v8f acc[3];
    #pragma unroll
    for (int s = 0; s < 3; ++s) acc[s] = (v8f){0.f,0.f,0.f,0.f,0.f,0.f,0.f,0.f};

    for (int tap = 0; tap < KTAP; ++tap) {
        int kd = tap / 9 - 1, kh = (tap / 3) % 3 - 1, kw = tap % 3 - 1;
        __syncthreads();
        // fill [n][c] tile: 64 rows x 16 chunks of 16B = 1024 chunks
        #pragma unroll
        for (int i = 0; i < 4; ++i) {
            int id = tid + (i << 8);
            int n = id >> 4, cc = id & 15;
            int v = vbase + n;
            int d = v >> 10, h = (v >> 5) & 31, w = v & 31;
            int dd = d + kd, hh = h + kh, ww = w + kw;
            unsigned lds = (unsigned)(size_t)&sBf[n * NPAD + cc * 4];
            if ((unsigned)dd < (unsigned)DD && (unsigned)hh < (unsigned)HH &&
                (unsigned)ww < (unsigned)WW) {
                int nv = (dd << 10) + (hh << 5) + ww;
                const float* g = t1 + ((size_t)(b * VOX + nv)) * CB + cc * 4;
                asm volatile("global_load_async_to_lds_b128 %0, %1, off"
                             :: "v"(lds), "v"((unsigned long long)(size_t)g)
                             : "memory");
            } else {
                float4* z = (float4*)&sBf[n * NPAD + cc * 4];
                *z = make_float4(0.f, 0.f, 0.f, 0.f);
            }
        }
        asm volatile("s_wait_asynccnt 0" ::: "memory");
        __syncthreads();
        #pragma unroll
        for (int q = 0; q < 2; ++q) {
            BF16x16 bh, bl;
            const float* rp = &sBf[nloc * NPAD + q * 32 + halfSel * 16];
            #pragma unroll
            for (int j = 0; j < 16; ++j) bfsplit(rp[j], bh.s[j], bl.s[j]);
            #pragma unroll
            for (int s = 0; s < 3; ++s) {
                int mi = (wv + s * 8) >> 2;
                int o = mi * 16 + lx;
                size_t ai = (((size_t)((tap * 2 + q) * 2 + halfSel)) * KOFFP + o) * 16;
                v16bf ah = ldFrag16(&wAH[ai]);
                v16bf al = ldFrag16(&wAL[ai]);
                acc[s] = wmma_bf16(ah, bh.v, acc[s]);
                acc[s] = wmma_bf16(ah, bl.v, acc[s]);
                acc[s] = wmma_bf16(al, bh.v, acc[s]);
            }
        }
    }
    #pragma unroll
    for (int s = 0; s < 3; ++s) {
        int mi = (wv + s * 8) >> 2;
        #pragma unroll
        for (int r = 0; r < 8; ++r) {
            int m = mi * 16 + r + halfSel * 8;
            float val = acc[s][r];
            if (m < KOFF) val += boff[m];
            offOut[((size_t)(b * KOFFP + m)) * VOX + vbase + nloc] = val;
        }
    }
}

// ---------------------------------------------------------------------------
// K5: fused deformable-sample GEMM + 1x1 conv + gating multiply.
//   acc tile split once at production into separate hi/lo ushort LDS tiles
//   [n][c] (pitch 72): every fragment is two conflict-free ds_load_b128,
//   zero unpack VALU. Weights: direct loads of pre-permuted fragments.
// ---------------------------------------------------------------------------
__global__ __launch_bounds__(256) void k_deform(const float* __restrict__ t1,
                                                const float* __restrict__ offIn,
                                                const unsigned short* __restrict__ wdAH,
                                                const unsigned short* __restrict__ wdAL,
                                                const float* __restrict__ bdef,
                                                const unsigned short* __restrict__ w1AH,
                                                const unsigned short* __restrict__ w1AL,
                                                const float* __restrict__ b1,
                                                const float* __restrict__ x,
                                                float* __restrict__ out) {
    __shared__ unsigned short sBH[CB * BP];   // [n][c] hi bf16
    __shared__ unsigned short sBL[CB * BP];   // [n][c] lo bf16
    __shared__ int   sCi[8][64];
    __shared__ float sCw[8][64];
    int tid = threadIdx.x;
    int lane = tid & 31, wv = tid >> 5;
    int halfSel = (lane >> 4) & 1;
    int lx = lane & 15;
    int b = blockIdx.x >> 8;
    int vbase = (blockIdx.x & 255) << 6;
    int mi = wv & 3;
    int nset = wv >> 2;
    int nloc0 = (nset * 2) * 16 + lx;
    int nloc1 = (nset * 2 + 1) * 16 + lx;
    int o = mi * 16 + lx;

    v8f acc0 = (v8f){0.f,0.f,0.f,0.f,0.f,0.f,0.f,0.f};
    v8f acc1 = (v8f){0.f,0.f,0.f,0.f,0.f,0.f,0.f,0.f};

    for (int tap = 0; tap < KTAP; ++tap) {
        __syncthreads();
        if (tid < 64) {
            int n = tid;
            int v = vbase + n;
            int d = v >> 10, h = (v >> 5) & 31, w = v & 31;
            size_t ob = ((size_t)(b * KOFFP + tap * 3)) * VOX + v;
            float od = offIn[ob];
            float oh = offIn[ob + (size_t)VOX];
            float ow = offIn[ob + (size_t)2 * VOX];
            float pd = (float)d + (float)(tap / 9 - 1) + od;
            float ph = (float)h + (float)((tap / 3) % 3 - 1) + oh;
            float pw = (float)w + (float)(tap % 3 - 1) + ow;
            float d0f = floorf(pd), h0f = floorf(ph), w0f = floorf(pw);
            float fd = pd - d0f, fh = ph - h0f, fw = pw - w0f;
            int d0 = (int)d0f, h0 = (int)h0f, w0 = (int)w0f;
            #pragma unroll
            for (int j = 0; j < 8; ++j) {
                int jd = (j >> 2) & 1, jh = (j >> 1) & 1, jw = j & 1;
                int di = d0 + jd, hi = h0 + jh, wi = w0 + jw;
                bool valid = (unsigned)di < (unsigned)DD &&
                             (unsigned)hi < (unsigned)HH &&
                             (unsigned)wi < (unsigned)WW;
                int dic = min(max(di, 0), DD - 1);
                int hic = min(max(hi, 0), HH - 1);
                int wic = min(max(wi, 0), WW - 1);
                sCi[j][n] = (dic << 10) + (hic << 5) + wic;
                float wt = (jd ? fd : 1.f - fd) * (jh ? fh : 1.f - fh) *
                           (jw ? fw : 1.f - fw);
                sCw[j][n] = valid ? wt : 0.f;
            }
        }
        __syncthreads();
        // gather: each thread does 4 float4 rows; split+store hi/lo tiles
        #pragma unroll
        for (int i = 0; i < 4; ++i) {
            int p = tid + (i << 8);
            int n = p >> 4;
            int c4 = (p & 15) << 2;
            const float4* basep =
                (const float4*)(t1 + ((size_t)b * VOX) * CB + c4);
            float sv[4] = {0.f, 0.f, 0.f, 0.f};
            #pragma unroll
            for (int j = 0; j < 8; ++j) {
                float wt = sCw[j][n];
                float4 vv = basep[(size_t)sCi[j][n] * 16];
                sv[0] += wt * vv.x; sv[1] += wt * vv.y;
                sv[2] += wt * vv.z; sv[3] += wt * vv.w;
            }
            u16x4 hv, lv;
            #pragma unroll
            for (int e = 0; e < 4; ++e) {
                unsigned short he, le;
                bfsplit(sv[e], he, le);
                hv[e] = he; lv[e] = le;
            }
            *(u16x4*)&sBH[n * BP + c4] = hv;
            *(u16x4*)&sBL[n * BP + c4] = lv;
        }
        __syncthreads();
        #pragma unroll
        for (int q = 0; q < 2; ++q) {
            int cb = q * 32 + halfSel * 16;
            v16bf b0h = ldFrag16(&sBH[nloc0 * BP + cb]);
            v16bf b0l = ldFrag16(&sBL[nloc0 * BP + cb]);
            v16bf b1h = ldFrag16(&sBH[nloc1 * BP + cb]);
            v16bf b1l = ldFrag16(&sBL[nloc1 * BP + cb]);
            size_t ai = (((size_t)((tap * 2 + q) * 2 + halfSel)) * CB + o) * 16;
            v16bf ah = ldFrag16(&wdAH[ai]);
            v16bf al = ldFrag16(&wdAL[ai]);
            acc0 = wmma_bf16(ah, b0h, acc0);
            acc0 = wmma_bf16(ah, b0l, acc0);
            acc0 = wmma_bf16(al, b0h, acc0);
            acc1 = wmma_bf16(ah, b1h, acc1);
            acc1 = wmma_bf16(ah, b1l, acc1);
            acc1 = wmma_bf16(al, b1h, acc1);
        }
    }

    // attn2 tile (+b_def) -> hi/lo LDS tiles [n][m] (8 contiguous m per lane)
    __syncthreads();
    {
        int mbase = mi * 16 + halfSel * 8;
        u16x8 h0v, l0v, h1v, l1v;
        #pragma unroll
        for (int r = 0; r < 8; ++r) {
            float bb = bdef[mbase + r];
            unsigned short he, le;
            bfsplit(acc0[r] + bb, he, le);
            h0v[r] = he; l0v[r] = le;
            bfsplit(acc1[r] + bb, he, le);
            h1v[r] = he; l1v[r] = le;
        }
        *(u16x8*)&sBH[nloc0 * BP + mbase] = h0v;
        *(u16x8*)&sBL[nloc0 * BP + mbase] = l0v;
        *(u16x8*)&sBH[nloc1 * BP + mbase] = h1v;
        *(u16x8*)&sBL[nloc1 * BP + mbase] = l1v;
    }
    __syncthreads();

    // 1x1 conv: attn3 = W1(64x64) @ attn2_tile + b1
    v8f o0 = (v8f){0.f,0.f,0.f,0.f,0.f,0.f,0.f,0.f};
    v8f o1 = (v8f){0.f,0.f,0.f,0.f,0.f,0.f,0.f,0.f};
    #pragma unroll
    for (int q = 0; q < 2; ++q) {
        int cb = q * 32 + halfSel * 16;
        v16bf b0h = ldFrag16(&sBH[nloc0 * BP + cb]);
        v16bf b0l = ldFrag16(&sBL[nloc0 * BP + cb]);
        v16bf b1h = ldFrag16(&sBH[nloc1 * BP + cb]);
        v16bf b1l = ldFrag16(&sBL[nloc1 * BP + cb]);
        size_t ai = ((size_t)(q * 2 + halfSel) * CB + o) * 16;
        v16bf ah = ldFrag16(&w1AH[ai]);
        v16bf al = ldFrag16(&w1AL[ai]);
        o0 = wmma_bf16(ah, b0h, o0);
        o0 = wmma_bf16(ah, b0l, o0);
        o0 = wmma_bf16(al, b0h, o0);
        o1 = wmma_bf16(ah, b1h, o1);
        o1 = wmma_bf16(ah, b1l, o1);
        o1 = wmma_bf16(al, b1h, o1);
    }
    #pragma unroll
    for (int r = 0; r < 8; ++r) {
        int m = mi * 16 + r + halfSel * 8;
        float bb = b1[m];
        size_t i0 = ((size_t)(b * CB + m)) * VOX + vbase + nloc0;
        size_t i1 = ((size_t)(b * CB + m)) * VOX + vbase + nloc1;
        out[i0] = (o0[r] + bb) * x[i0];
        out[i1] = (o1[r] + bb) * x[i1];
    }
}

// ---------------------------------------------------------------------------
// Host launcher
// ---------------------------------------------------------------------------
extern "C" void kernel_launch(void* const* d_in, const int* in_sizes, int n_in,
                              void* d_out, int out_size, void* d_ws, size_t ws_size,
                              hipStream_t stream) {
    const float* x     = (const float*)d_in[0];
    const float* w0    = (const float*)d_in[1];
    const float* b0    = (const float*)d_in[2];
    const float* wsW   = (const float*)d_in[3];
    const float* bs    = (const float*)d_in[4];
    const float* w_off = (const float*)d_in[5];
    const float* b_off = (const float*)d_in[6];
    const float* w_def = (const float*)d_in[7];
    const float* b_def = (const float*)d_in[8];
    const float* w1    = (const float*)d_in[9];
    const float* b1    = (const float*)d_in[10];
    float* out = (float*)d_out;

    const size_t NCV = (size_t)NB * VOX * CB;      // 2,097,152 floats
    float* x_t  = (float*)d_ws;
    float* t0   = x_t + NCV;
    float* t1   = t0 + NCV;
    float* offb = t1 + NCV;                        // [b][96][VOX]
    unsigned short* woAH = (unsigned short*)(offb + (size_t)NB * KOFFP * VOX);
    unsigned short* woAL = woAH + WOFF_N;
    unsigned short* wdAH = woAL + WOFF_N;
    unsigned short* wdAL = wdAH + WDEF_N;
    unsigned short* w1AH = wdAL + WDEF_N;
    unsigned short* w1AL = w1AH + W1_N;

    const int nElem = (int)NCV;
    k_transpose<<<nElem / 256, 256, 0, stream>>>(x, x_t);
    k_wprep<<<(WOFF_N + WDEF_N + W1_N + 255) / 256, 256, 0, stream>>>(
        w_off, w_def, w1, woAH, woAL, wdAH, wdAL, w1AH, w1AL);
    k_dw5<<<nElem / 256, 256, 0, stream>>>(x_t, w0, b0, t0);
    k_dw7d3<<<nElem / 256, 256, 0, stream>>>(t0, wsW, bs, t1);
    k_offconv<<<NB * (VOX / 64), 256, 0, stream>>>(t1, woAH, woAL, b_off, offb);
    k_deform<<<NB * (VOX / 64), 256, 0, stream>>>(t1, offb, wdAH, wdAL, b_def,
                                                  w1AH, w1AL, b1, x, out);
    (void)in_sizes; (void)n_in; (void)out_size; (void)ws_size;
}
